// CrossAttention_3470333575276
// MI455X (gfx1250) — compile-verified, matrix operand order checked by reading, not run on hardware
//
#include <hip/hip_runtime.h>
#include <hip/hip_bf16.h>

typedef __attribute__((ext_vector_type(16))) __bf16 v16bf;
typedef __attribute__((ext_vector_type(8)))  float  v8f;

#define NHEADS   16
#define HDIM     64
#define MODELDIM 1024
#define SRCDIM   1152

#if defined(__has_builtin)
#if __has_builtin(__builtin_amdgcn_tensor_load_to_lds)
#define HAVE_TDM 1
#endif
#endif

// K-index pattern for 16-bit A fragments (ISA 7.12.2): lane half h, VGPR j
// j 0..3  -> K = 8h + 2j, 8h + 2j + 1
// j 4..7  -> K = 16 + 8h + 2(j-4), +1
__device__ __forceinline__ int kpat(int h, int j) {
  return (j < 4) ? (8 * h + 2 * j) : (16 + 8 * h + 2 * (j - 4));
}

union U32B2 { unsigned int u; unsigned short s[2]; };

__device__ __forceinline__ __bf16 us_to_bf(unsigned short u) {
  __bf16 r; __builtin_memcpy(&r, &u, 2); return r;
}

__device__ __forceinline__ v8f wmma_bf16(v16bf a, v16bf b, v8f c) {
  return __builtin_amdgcn_wmma_f32_16x16x32_bf16(false, a, false, b, (short)0, c,
                                                 false, false);
}

// Load a 16x32 A-style fragment (also used for B as A-layout of B^T) from a
// row-major fp32 matrix, converting to bf16 in registers. `row` already
// includes the per-lane row (base + lane%16); h = lane/16.
__device__ __forceinline__ v16bf frag_from_f32(const float* __restrict__ p,
                                               int ld, int row, int k0, int h) {
  const float* base = p + (size_t)row * ld + k0;
  v16bf f = {};
#pragma unroll
  for (int j = 0; j < 8; ++j) {
    int k = kpat(h, j);
    float2 v = *(const float2*)(base + k);   // k even -> 8B aligned
    f[2 * j]     = (__bf16)v.x;
    f[2 * j + 1] = (__bf16)v.y;
  }
  return f;
}

__device__ __forceinline__ v16bf frag_from_bf16(const __bf16* __restrict__ p,
                                                int ld, int row, int k0, int h) {
  const __bf16* base = p + (size_t)row * ld + k0;
  v16bf f = {};
#pragma unroll
  for (int j = 0; j < 8; ++j) {
    int k = kpat(h, j);
    U32B2 u; u.u = *(const unsigned int*)(base + k);  // 2 bf16 per dword
    f[2 * j]     = us_to_bf(u.s[0]);
    f[2 * j + 1] = us_to_bf(u.s[1]);
  }
  return f;
}

#ifdef HAVE_TDM
typedef unsigned int v4u  __attribute__((ext_vector_type(4)));
typedef int          v8it __attribute__((ext_vector_type(8)));
typedef int          v4it __attribute__((ext_vector_type(4)));

// TDM: DMA one 32-row x 64-col bf16 tile (row-major, stride 64) from global
// into LDS. D# encoding per CDNA5 ISA 8.3/8.4:
//  group0: count=1 | lds_addr | global_addr[56:0] | type=2
//  group1: data_size=1 (2B), tensor_dim0=64, tensor_dim1=32, tile=64x32,
//          tensor_dim0_stride=64; groups 2/3 (and trailing group) zero for a
//          2-D tile.
// Wave-level op (EXEC ignored); tracked by TENSORcnt.
// This toolchain's builtin is the 6-arg clang-23 form:
//   (uint32x4 g0, int32x8 g1, int32x4, int32x4, int32x8, i32 cpol)
__device__ __forceinline__ void tdm_load_tile_32x64_bf16(const __bf16* gptr,
                                                         void* lds_ptr) {
  unsigned lds = (unsigned)(size_t)lds_ptr;  // low 32b of generic addr = LDS byte addr
  unsigned long long ga = (unsigned long long)(size_t)gptr;
  v4u g0;
  g0[0] = 1u;                                            // count = 1 (valid)
  g0[1] = lds;                                           // lds_addr
  g0[2] = (unsigned)(ga & 0xffffffffu);                  // global_addr[31:0]
  g0[3] = (unsigned)((ga >> 32) & 0x1ffffffu) | (2u << 30);  // ga[56:32] | type=2
  v8it g1;
  g1[0] = (int)(1u << 16);    // data_size = 1 -> 2 bytes/elem
  g1[1] = (int)(64u << 16);   // tensor_dim0 = 64
  g1[2] = (int)(32u << 16);   // tensor_dim1 = 32
  g1[3] = (int)(64u << 16);   // tile_dim0 = 64
  g1[4] = 32;                 // tile_dim1 = 32, tile_dim2 = 0
  g1[5] = 64;                 // tensor_dim0_stride = 64 elems
  g1[6] = 0;
  g1[7] = 0;
  v4it z4 = {};
  v8it z8 = {};
  __builtin_amdgcn_tensor_load_to_lds(g0, g1, z4, z4, z8, 0);
}
#endif

// Generic projection GEMM: out = src @ W^T + bias.
//  src: fp32 [N,C] (Xf) or bf16 [N,C] (Xb)  — exactly one non-null.
//  mode fp32 (outF != null): write fp32 row-major [N, MODELDIM]   (final proj)
//  mode bf16 (outB != null): write bf16 [B, NHEADS, seq, HDIM], optional RoPE.
__global__ __launch_bounds__(256) void proj_gemm_kernel(
    const float* __restrict__ Xf, const __bf16* __restrict__ Xb,
    const float* __restrict__ W, const float* __restrict__ bias,
    float* __restrict__ outF, __bf16* __restrict__ outB,
    int C, int seq, int do_rope) {
  const int w    = threadIdx.x >> 5;
  const int lane = threadIdx.x & 31;
  const int h2   = lane >> 4;
  const int n    = lane & 15;
  const int r0   = blockIdx.y * 128 + w * 16;
  const int c0   = blockIdx.x * 64;

  v8f acc[4] = {};
  for (int kk = 0; kk < C; kk += 32) {
    v16bf a = Xf ? frag_from_f32(Xf, C, r0 + n, kk, h2)
                 : frag_from_bf16(Xb, C, r0 + n, kk, h2);
#pragma unroll
    for (int s = 0; s < 4; ++s) {
      v16bf b = frag_from_f32(W, C, c0 + 16 * s + n, kk, h2);
      acc[s] = wmma_bf16(a, b, acc[s]);
    }
  }

  // C/D layout: lane holds column c0+16s+n, rows r + 8*h2 for r=0..7.
#pragma unroll
  for (int s = 0; s < 4; ++s) {
    const int c   = c0 + 16 * s + n;
    const float bv = bias[c];
    if (outF) {
#pragma unroll
      for (int r = 0; r < 8; ++r) {
        int grow = r0 + r + 8 * h2;
        outF[(size_t)grow * MODELDIM + c] = acc[s][r] + bv;
      }
    } else {
      const int head = c >> 6;
      const int d    = c & 63;
      // NeoX tables: cos/sin index uses d&31; rotation pairs (2i, 2i+1).
      const float sgn = (d & 1) ? 1.0f : -1.0f;
      float inv = 0.0f;
      if (do_rope)
        inv = __expf(-(float)(2 * (d & 31)) * (9.210340371976184f / 64.0f));
#pragma unroll
      for (int r = 0; r < 8; ++r) {
        int grow = r0 + r + 8 * h2;
        int bidx = grow / seq;
        int t    = grow % seq;
        float val = acc[s][r] + bv;
        if (do_rope) {
          float pv = __shfl_xor(val, 1);  // partner lane = other half of pair
          float sn, cs;
          __sincosf((float)t * inv, &sn, &cs);
          val = val * cs + sgn * pv * sn;
        }
        outB[(((size_t)bidx * NHEADS + head) * seq + t) * HDIM + d] = (__bf16)val;
      }
    }
  }
}

// Flash cross-attention for one (b, h): 8 waves x 16 query rows = 128 queries
// per block. K/V streamed in 32-key chunks into double-buffered LDS via the
// Tensor Data Mover (overlapping DMA of chunk i+1 with compute of chunk i);
// online softmax in f32, all GEMMs on the WMMA pipe.
__global__ __launch_bounds__(256) void attn_kernel(
    const __bf16* __restrict__ Q, const __bf16* __restrict__ Kp,
    const __bf16* __restrict__ Vp, __bf16* __restrict__ O, int T, int Kc) {
  __shared__ __bf16 sK[2][32 * 64];
  __shared__ __bf16 sV[2][32 * 64];
  __shared__ __bf16 sP[8][16 * 32];  // per-wave P transpose staging

  const int b = blockIdx.z, h = blockIdx.y;
  const int w = threadIdx.x >> 5, lane = threadIdx.x & 31;
  const int hh = lane >> 4, n = lane & 15;
  const size_t bh = (size_t)b * NHEADS + h;
  const __bf16* Qbh = Q + bh * (size_t)T * HDIM;
  const __bf16* Kbh = Kp + bh * (size_t)Kc * HDIM;
  const __bf16* Vbh = Vp + bh * (size_t)Kc * HDIM;
  const int q0 = blockIdx.x * 128 + w * 16;

  // Q fragments for the two 32-wide d chunks, resident for the whole pass.
  v16bf qf0 = frag_from_bf16(Qbh, HDIM, q0 + n, 0, hh);
  v16bf qf1 = frag_from_bf16(Qbh, HDIM, q0 + n, 32, hh);

  v8f o[4] = {};
  float mrow[8], lrow[8];
#pragma unroll
  for (int r = 0; r < 8; ++r) { mrow[r] = -1e30f; lrow[r] = 0.0f; }
  const float scale = 0.125f;  // 1/sqrt(64)

  const int nch = Kc / 32;
#ifdef HAVE_TDM
  if (w == 0) {  // wave-level TDM issue for chunk 0
    tdm_load_tile_32x64_bf16(Kbh, sK[0]);
    tdm_load_tile_32x64_bf16(Vbh, sV[0]);
  }
#endif

  for (int ic = 0; ic < nch; ++ic) {
    const int buf = ic & 1;
#ifdef HAVE_TDM
    if (w == 0) __builtin_amdgcn_s_wait_tensorcnt(0);
    __syncthreads();  // chunk ic resident in sK[buf]/sV[buf] for all waves
    if (w == 0 && ic + 1 < nch) {  // prefetch chunk ic+1 into the other buffer
      tdm_load_tile_32x64_bf16(Kbh + (size_t)(ic + 1) * 32 * HDIM, sK[buf ^ 1]);
      tdm_load_tile_32x64_bf16(Vbh + (size_t)(ic + 1) * 32 * HDIM, sV[buf ^ 1]);
    }
#else
    {
      const unsigned int* gk = (const unsigned int*)(Kbh + (size_t)ic * 32 * HDIM);
      const unsigned int* gv = (const unsigned int*)(Vbh + (size_t)ic * 32 * HDIM);
      unsigned int* sk = (unsigned int*)sK[buf];
      unsigned int* sv = (unsigned int*)sV[buf];
      for (int i = threadIdx.x; i < 1024; i += 256) { sk[i] = gk[i]; sv[i] = gv[i]; }
    }
    __syncthreads();
#endif

    // Scores: 16 queries x 32 keys = two 16x16 WMMA accumulators.
    v8f s0 = {}, s1 = {};
#pragma unroll
    for (int dc = 0; dc < 2; ++dc) {
      const int d0 = dc * 32;
      v16bf bk0 = {}, bk1 = {};
#pragma unroll
      for (int j = 0; j < 8; ++j) {
        int k = kpat(hh, j);
        U32B2 u0; u0.u = *(const unsigned int*)(sK[buf] + (0 + n) * 64 + d0 + k);
        U32B2 u1; u1.u = *(const unsigned int*)(sK[buf] + (16 + n) * 64 + d0 + k);
        bk0[2 * j] = us_to_bf(u0.s[0]); bk0[2 * j + 1] = us_to_bf(u0.s[1]);
        bk1[2 * j] = us_to_bf(u1.s[0]); bk1[2 * j + 1] = us_to_bf(u1.s[1]);
      }
      s0 = wmma_bf16(dc ? qf1 : qf0, bk0, s0);
      s1 = wmma_bf16(dc ? qf1 : qf0, bk1, s1);
    }

    // Online softmax. Each lane owns column n (and 16+n), rows r+8*hh.
    // Row reductions: shfl_xor masks 1..8 stay inside each 16-lane half.
#pragma unroll
    for (int r = 0; r < 8; ++r) {
      float a0 = s0[r] * scale, a1 = s1[r] * scale;
      float x = fmaxf(a0, a1);
#pragma unroll
      for (int off = 8; off >= 1; off >>= 1) x = fmaxf(x, __shfl_xor(x, off));
      float mn = fmaxf(mrow[r], x);
      float alpha = __expf(mrow[r] - mn);
      mrow[r] = mn;
      float p0 = __expf(a0 - mn), p1 = __expf(a1 - mn);
      float rs = p0 + p1;
#pragma unroll
      for (int off = 8; off >= 1; off >>= 1) rs += __shfl_xor(rs, off);
      lrow[r] = lrow[r] * alpha + rs;
#pragma unroll
      for (int s = 0; s < 4; ++s) o[s][r] *= alpha;
      // C-layout -> row-major 16x32 P tile in per-wave LDS (in-wave DS order).
      sP[w][(r + 8 * hh) * 32 + n]      = (__bf16)p0;
      sP[w][(r + 8 * hh) * 32 + 16 + n] = (__bf16)p1;
    }

    // P as A-fragment (16 queries x 32 keys).
    v16bf pf = {};
#pragma unroll
    for (int j = 0; j < 8; ++j) {
      int k = kpat(hh, j);
      U32B2 u; u.u = *(const unsigned int*)(&sP[w][n * 32 + k]);
      pf[2 * j] = us_to_bf(u.s[0]); pf[2 * j + 1] = us_to_bf(u.s[1]);
    }
    // V as B-fragments, one per 16-wide d subtile; K-dim = 32 keys.
#pragma unroll
    for (int s = 0; s < 4; ++s) {
      v16bf vf = {};
#pragma unroll
      for (int j = 0; j < 8; ++j) {
        int k = kpat(hh, j);
        vf[2 * j]     = sV[buf][k * 64 + 16 * s + n];
        vf[2 * j + 1] = sV[buf][(k + 1) * 64 + 16 * s + n];
      }
      o[s] = wmma_bf16(pf, vf, o[s]);
    }
    __syncthreads();  // all waves done with sK[buf]/sV[buf] before refill
  }

  // Normalize and write bf16 [B, T, 1024] for the output projection.
#pragma unroll
  for (int s = 0; s < 4; ++s) {
#pragma unroll
    for (int r = 0; r < 8; ++r) {
      int t = q0 + r + 8 * hh;
      float val = o[s][r] / lrow[r];
      O[((size_t)b * T + t) * MODELDIM + h * HDIM + 16 * s + n] = (__bf16)val;
    }
  }
}

extern "C" void kernel_launch(void* const* d_in, const int* in_sizes, int n_in,
                              void* d_out, int out_size, void* d_ws, size_t ws_size,
                              hipStream_t stream) {
  (void)in_sizes; (void)n_in; (void)out_size; (void)ws_size;
  const float* x   = (const float*)d_in[0];
  const float* ctx = (const float*)d_in[1];
  const float* Wq  = (const float*)d_in[2];
  const float* bq  = (const float*)d_in[3];
  const float* Wk  = (const float*)d_in[4];
  const float* bk  = (const float*)d_in[5];
  const float* Wv  = (const float*)d_in[6];
  const float* bv  = (const float*)d_in[7];
  const float* Wo  = (const float*)d_in[8];
  const float* bo  = (const float*)d_in[9];
  float* out = (float*)d_out;

  const int B = 2, T = 2048, Kc = 2048;
  // Workspace: Q, K, V as bf16 [B,H,seq,64] (8 MB each) + attn out bf16
  // [B,T,1024] (8 MB) = 32 MB total.
  const size_t elems = (size_t)B * NHEADS * T * HDIM;  // 4,194,304
  char* ws = (char*)d_ws;
  __bf16* wQ = (__bf16*)(ws);
  __bf16* wK = (__bf16*)(ws + elems * 2);
  __bf16* wV = (__bf16*)(ws + elems * 4);
  __bf16* wO = (__bf16*)(ws + elems * 6);

  dim3 blk(256);
  // Q = x @ Wq^T + bq, RoPE, -> bf16 [B,H,T,64]
  proj_gemm_kernel<<<dim3(MODELDIM / 64, (B * T) / 128), blk, 0, stream>>>(
      x, nullptr, Wq, bq, nullptr, wQ, MODELDIM, T, 1);
  // K = ctx @ Wk^T + bk, RoPE
  proj_gemm_kernel<<<dim3(MODELDIM / 64, (B * Kc) / 128), blk, 0, stream>>>(
      ctx, nullptr, Wk, bk, nullptr, wK, SRCDIM, Kc, 1);
  // V = ctx @ Wv^T + bv
  proj_gemm_kernel<<<dim3(MODELDIM / 64, (B * Kc) / 128), blk, 0, stream>>>(
      ctx, nullptr, Wv, bv, nullptr, wV, SRCDIM, Kc, 0);
  // Flash cross-attention -> bf16 [B,T,1024]
  attn_kernel<<<dim3(T / 128, NHEADS, B), blk, 0, stream>>>(wQ, wK, wV, wO, T, Kc);
  // out = attn @ Wo^T + bo -> fp32 d_out
  proj_gemm_kernel<<<dim3(MODELDIM / 64, (B * T) / 128), blk, 0, stream>>>(
      nullptr, wO, Wo, bo, out, nullptr, MODELDIM, T, 0);
}
